// GADGCNN_32298154066800
// MI455X (gfx1250) — compile-verified, matrix operand order checked by reading, not run on hardware
//
#include <hip/hip_runtime.h>

// ---------------------------------------------------------------------------
// DGCNN segmentation head for MI455X (gfx1250): bf16 WMMA GEMMs for all 1x1
// convs, pos-major bf16 activations so WMMA B-fragments / D-tiles map to
// contiguous 16B global loads/stores, register top-16 KNN with LDS staging.
// GEMM K-chunk is load-phase/compute-phase split so the wave issues all 10
// b128 loads as one clause and drains the load counter incrementally across
// the 4 WMMAs instead of fully stalling before each one.
// ---------------------------------------------------------------------------

namespace {

constexpr int NPT = 4096;
constexpr int NBATCH = 8;
constexpr int KNB = 16;
constexpr int P1 = NBATCH * NPT;   // 32768 points
constexpr int P2 = P1 * KNB;       // 524288 point-neighbor pairs
constexpr int LD3 = 1184;          // concat arena row stride (halves)
constexpr int CH_F2 = 32;          // fea2 channel offset in arena
constexpr int CH_G = 160;          // global-feature channel offset in arena

typedef unsigned short u16;
typedef unsigned int u32;
typedef __attribute__((ext_vector_type(16))) __bf16 v16bf;
typedef __attribute__((ext_vector_type(8))) float v8f;

union FragAB { v16bf v; uint4 q[2]; };

__device__ __forceinline__ u16 f2b(float x) {          // RNE float->bf16
  u32 u = __float_as_uint(x);
  u32 r = u + 0x7fffu + ((u >> 16) & 1u);
  return (u16)(r >> 16);
}
__device__ __forceinline__ float b2f(u16 h) { return __uint_as_float(((u32)h) << 16); }
__device__ __forceinline__ u32 pk2(float a, float b) {
  return (u32)f2b(a) | ((u32)f2b(b) << 16);
}

// ---------------------------------------------------------------------------
// Generic bf16 WMMA GEMM: Y[p, m] (pos-major bf16, row stride ldy) or final
// fp32 (B, CO, N).  W: (co_pad, CIP) bf16 row-major.  X: pos-major, stride ldx.
// One wave: 16 output channels x 64 positions. Block: 8 waves (512 positions).
// ---------------------------------------------------------------------------
__global__ __launch_bounds__(256) void gemm_bf16(
    const u16* __restrict__ W, const float* __restrict__ bias,
    const u16* __restrict__ X, int ldx,
    u16* __restrict__ Y, int ldy,
    float* __restrict__ Yf,
    int CO, int CIP, int relu) {
  const int lane = threadIdx.x & 31;
  const int wv = threadIdx.x >> 5;
  const int m0 = blockIdx.y << 4;
  const int t0 = (blockIdx.x * 8 + wv) << 2;   // first 16-wide position tile
  const int row = lane & 15;                   // M (for A) / N (for B/D)
  const int hb = (lane >> 4) << 3;             // half-K base: 0 or 8

  v8f acc[4];
#pragma unroll
  for (int i = 0; i < 4; ++i)
#pragma unroll
    for (int j = 0; j < 8; ++j) acc[i][j] = 0.f;

  const u16* wp = W + (size_t)(m0 + row) * CIP + hb;
  const u16* xp[4];
#pragma unroll
  for (int nt = 0; nt < 4; ++nt)
    xp[nt] = X + (size_t)((t0 + nt) * 16 + row) * ldx + hb;

  for (int kc = 0; kc < CIP; kc += 32) {
    // ---- load phase: A fragment + 4 B fragments, issued as one batch ----
    FragAB a, bf[4];
    a.q[0] = *(const uint4*)(wp + kc);
    a.q[1] = *(const uint4*)(wp + kc + 16);
#pragma unroll
    for (int nt = 0; nt < 4; ++nt) {
      bf[nt].q[0] = *(const uint4*)(xp[nt] + kc);
      bf[nt].q[1] = *(const uint4*)(xp[nt] + kc + 16);
    }
    // ---- compute phase: 4 back-to-back WMMAs reusing the A fragment ----
#pragma unroll
    for (int nt = 0; nt < 4; ++nt)
      acc[nt] = __builtin_amdgcn_wmma_f32_16x16x32_bf16(
          false, a.v, false, bf[nt].v, (short)0, acc[nt], false, false);
  }

  // D layout: lane holds N = row, M = m0 + hb + r  (r = 0..7, contiguous)
  float bv[8];
#pragma unroll
  for (int r = 0; r < 8; ++r) {
    int m = m0 + hb + r;
    bv[r] = (m < CO) ? bias[m] : 0.f;
  }

#pragma unroll
  for (int nt = 0; nt < 4; ++nt) {
    const int p = (t0 + nt) * 16 + row;
    if (Yf) {  // final layer: fp32 output, layout (B, CO, N)
#pragma unroll
      for (int r = 0; r < 8; ++r) {
        int m = m0 + hb + r;
        float v = acc[nt][r] + bv[r];
        if (relu) v = fmaxf(v, 0.f);
        if (m < CO) {
          int b = p >> 12, n = p & (NPT - 1);
          Yf[((size_t)b * CO + m) * NPT + n] = v;
        }
      }
    } else {   // chain layer: 8 bf16 = one contiguous 16B store (pos-major)
      u32 w4[4];
#pragma unroll
      for (int r = 0; r < 4; ++r) {
        float v0 = acc[nt][2 * r] + bv[2 * r];
        float v1 = acc[nt][2 * r + 1] + bv[2 * r + 1];
        if (relu) { v0 = fmaxf(v0, 0.f); v1 = fmaxf(v1, 0.f); }
        w4[r] = pk2(v0, v1);
      }
      uint4 o; o.x = w4[0]; o.y = w4[1]; o.z = w4[2]; o.w = w4[3];
      *(uint4*)(Y + (size_t)p * ldy + m0 + hb) = o;
    }
  }
}

// ---------------------------------------------------------------------------
// KNN: per-thread sorted top-16 (descending neg-dist; strict > keeps the
// earliest index on ties, matching lax.top_k; self-distance 0 is included).
// ---------------------------------------------------------------------------
#define KNN_INSERT(ndv, ci)                                                  \
  do {                                                                       \
    if ((ndv) > best[15]) {                                                  \
      float v_ = (ndv); int vi_ = (ci);                                      \
      _Pragma("unroll")                                                      \
      for (int t_ = 0; t_ < 16; ++t_) {                                      \
        if (v_ > best[t_]) {                                                 \
          float tv_ = best[t_]; int ti_ = bidx[t_];                          \
          best[t_] = v_; bidx[t_] = vi_; v_ = tv_; vi_ = ti_;                \
        }                                                                    \
      }                                                                      \
    }                                                                        \
  } while (0)

__global__ __launch_bounds__(256) void knn_xyz(const float* __restrict__ xyz,
                                               int* __restrict__ idx) {
  const int b = blockIdx.y;
  const int n = blockIdx.x * 256 + threadIdx.x;
  const float* q = xyz + ((size_t)b * NPT + n) * 3;
  const float qx = q[0], qy = q[1], qz = q[2];
  __shared__ float sc[256 * 3];
  float best[KNB]; int bidx[KNB];
#pragma unroll
  for (int t = 0; t < KNB; ++t) { best[t] = -3.4e38f; bidx[t] = 0; }
  for (int c0 = 0; c0 < NPT; c0 += 256) {
    __syncthreads();
    for (int t = threadIdx.x; t < 768; t += 256)
      sc[t] = xyz[(size_t)b * NPT * 3 + (size_t)c0 * 3 + t];
    __syncthreads();
    for (int c = 0; c < 256; ++c) {  // same c for all lanes: LDS broadcast
      float dx = qx - sc[c * 3 + 0];
      float dy = qy - sc[c * 3 + 1];
      float dz = qz - sc[c * 3 + 2];
      float nd = -(dx * dx + dy * dy + dz * dz);
      KNN_INSERT(nd, c0 + c);
    }
  }
  int* op = idx + ((size_t)b * NPT + n) * KNB;
#pragma unroll
  for (int t = 0; t < KNB; ++t) op[t] = bidx[t];
}

__global__ __launch_bounds__(256) void knn_fea(const u16* __restrict__ X3,
                                               int* __restrict__ idx) {
  const int b = blockIdx.y;
  const int n = blockIdx.x * 256 + threadIdx.x;
  const u32* qr = (const u32*)(X3 + ((size_t)b * NPT + n) * LD3);
  float qf[32];
#pragma unroll
  for (int d = 0; d < 16; ++d) {
    u32 u = qr[d];
    qf[2 * d] = __uint_as_float(u << 16);
    qf[2 * d + 1] = __uint_as_float(u & 0xffff0000u);
  }
  __shared__ u32 sc[128 * 16];  // 128 candidates x 32 bf16 channels
  float best[KNB]; int bidx[KNB];
#pragma unroll
  for (int t = 0; t < KNB; ++t) { best[t] = -3.4e38f; bidx[t] = 0; }
  for (int c0 = 0; c0 < NPT; c0 += 128) {
    __syncthreads();
    for (int t = threadIdx.x; t < 2048; t += 256) {
      int c = t >> 4, dw = t & 15;
      sc[t] = ((const u32*)(X3 + ((size_t)b * NPT + c0 + c) * LD3))[dw];
    }
    __syncthreads();
    for (int c = 0; c < 128; ++c) {
      const u32* cr = &sc[c << 4];
      float s = 0.f;
#pragma unroll
      for (int d = 0; d < 16; ++d) {
        u32 u = cr[d];
        float d0 = qf[2 * d] - __uint_as_float(u << 16);
        float d1 = qf[2 * d + 1] - __uint_as_float(u & 0xffff0000u);
        s = fmaf(d0, d0, fmaf(d1, d1, s));
      }
      float nd = -s;
      KNN_INSERT(nd, c0 + c);
    }
  }
  int* op = idx + ((size_t)b * NPT + n) * KNB;
#pragma unroll
  for (int t = 0; t < KNB; ++t) op[t] = bidx[t];
}

// ---------------------------------------------------------------------------
// Edgeconv input builders (bf16, pos-major, K-padded to 32/64 with zeros)
// ---------------------------------------------------------------------------
__global__ __launch_bounds__(256) void build_g(const float* __restrict__ xyz,
                                               const int* __restrict__ idx,
                                               u16* __restrict__ G) {
  const size_t pos2 = (size_t)blockIdx.x * 256 + threadIdx.x;
  const int kk = (int)(pos2 & (KNB - 1));
  const size_t pn = pos2 >> 4;
  const int b = (int)(pn >> 12), n = (int)(pn & (NPT - 1));
  const int j = idx[pn * KNB + kk];
  const float* pc = xyz + ((size_t)b * NPT + n) * 3;
  const float* pk = xyz + ((size_t)b * NPT + j) * 3;
  float cx = pc[0], cy = pc[1], cz = pc[2];
  float kx = pk[0], ky = pk[1], kz = pk[2];
  float rx = cx - kx, ry = cy - ky, rz = cz - kz;
  float d = sqrtf(rx * rx + ry * ry + rz * rz);
  union { u16 u[32]; uint4 q[4]; } r;
#pragma unroll
  for (int i = 0; i < 32; ++i) r.u[i] = 0;
  r.u[0] = f2b(d);
  r.u[1] = f2b(cx); r.u[2] = f2b(cy); r.u[3] = f2b(cz);
  r.u[4] = f2b(kx); r.u[5] = f2b(ky); r.u[6] = f2b(kz);
  r.u[7] = f2b(rx); r.u[8] = f2b(ry); r.u[9] = f2b(rz);
  uint4* o = (uint4*)(G + pos2 * 32);
#pragma unroll
  for (int i = 0; i < 4; ++i) o[i] = r.q[i];
}

__global__ __launch_bounds__(256) void build_f_xyz(const float* __restrict__ xyz,
                                                   const int* __restrict__ idx,
                                                   u16* __restrict__ F) {
  const size_t pos2 = (size_t)blockIdx.x * 256 + threadIdx.x;
  const int kk = (int)(pos2 & (KNB - 1));
  const size_t pn = pos2 >> 4;
  const int b = (int)(pn >> 12), n = (int)(pn & (NPT - 1));
  const int j = idx[pn * KNB + kk];
  const float* pc = xyz + ((size_t)b * NPT + n) * 3;
  const float* pk = xyz + ((size_t)b * NPT + j) * 3;
  union { u16 u[32]; uint4 q[4]; } r;
#pragma unroll
  for (int i = 0; i < 32; ++i) r.u[i] = 0;
  r.u[0] = f2b(pc[0]); r.u[1] = f2b(pc[1]); r.u[2] = f2b(pc[2]);
  r.u[3] = f2b(pk[0]); r.u[4] = f2b(pk[1]); r.u[5] = f2b(pk[2]);
  uint4* o = (uint4*)(F + pos2 * 32);
#pragma unroll
  for (int i = 0; i < 4; ++i) o[i] = r.q[i];
}

__global__ __launch_bounds__(256) void build_f_fea(const u16* __restrict__ X3,
                                                   const int* __restrict__ idx,
                                                   u16* __restrict__ F) {
  const size_t pos2 = (size_t)blockIdx.x * 256 + threadIdx.x;
  const int kk = (int)(pos2 & (KNB - 1));
  const size_t pn = pos2 >> 4;
  const int b = (int)(pn >> 12);
  const int j = idx[pn * KNB + kk];
  const uint4* fc = (const uint4*)(X3 + pn * LD3);                         // 32ch center
  const uint4* fk = (const uint4*)(X3 + ((size_t)b * NPT + j) * LD3);     // 32ch neighbor
  uint4* o = (uint4*)(F + pos2 * 64);
#pragma unroll
  for (int i = 0; i < 4; ++i) o[i] = fc[i];
#pragma unroll
  for (int i = 0; i < 4; ++i) o[4 + i] = fk[i];
}

// Epilogue: fea_out[pn, co] = max_k( g * f ), written into concat arena (bf16)
__global__ __launch_bounds__(256) void ec_mul_max(const u16* __restrict__ G,
                                                  const u16* __restrict__ F,
                                                  int ld, int Co,
                                                  u16* __restrict__ X3, int choff) {
  const size_t g = (size_t)blockIdx.x * 256 + threadIdx.x;
  const int co = (int)(g % Co);
  const size_t pn = g / Co;
  float mx = -3.4e38f;
#pragma unroll 4
  for (int kk = 0; kk < KNB; ++kk) {
    size_t base = (pn * KNB + kk) * (size_t)ld + co;
    mx = fmaxf(mx, b2f(G[base]) * b2f(F[base]));
  }
  X3[pn * LD3 + choff + co] = f2b(mx);
}

// Global max over N of c2 output (pos-major bf16, 1024 ch): two phases
__global__ __launch_bounds__(256) void gmax_part(const u16* __restrict__ Y2,
                                                 float* __restrict__ part) {
  const int co = blockIdx.x * 256 + threadIdx.x;   // gridDim.x = 4
  const int seg = blockIdx.y;                      // 8 segments of 512
  const int b = blockIdx.z;
  float mx = -3.4e38f;
  for (int n = seg * 512; n < seg * 512 + 512; ++n)
    mx = fmaxf(mx, b2f(Y2[((size_t)b * NPT + n) * 1024 + co]));
  part[((size_t)b * 8 + seg) * 1024 + co] = mx;
}

__global__ __launch_bounds__(256) void gmax_bcast(const float* __restrict__ part,
                                                  u16* __restrict__ X3) {
  const int co = blockIdx.x * 256 + threadIdx.x;   // gridDim.x = 4
  const int n0 = blockIdx.y * 256;                 // 16 n-chunks
  const int b = blockIdx.z;
  float mx = -3.4e38f;
#pragma unroll
  for (int s = 0; s < 8; ++s)
    mx = fmaxf(mx, part[((size_t)b * 8 + s) * 1024 + co]);
  const u16 h = f2b(mx);
  for (int n = n0; n < n0 + 256; ++n)
    X3[((size_t)b * NPT + n) * LD3 + CH_G + co] = h;
}

// fp32 weight (co, ci) -> zero-padded bf16 (co_pad, ci_pad)
__global__ __launch_bounds__(256) void cvt_w(const float* __restrict__ w,
                                             u16* __restrict__ wp,
                                             int co, int ci, int ci_pad, int total) {
  const int g = blockIdx.x * 256 + threadIdx.x;
  if (g >= total) return;
  const int o = g / ci_pad, i = g % ci_pad;
  const float v = (o < co && i < ci) ? w[(size_t)o * ci + i] : 0.f;
  wp[g] = f2b(v);
}

}  // namespace

// ---------------------------------------------------------------------------
extern "C" void kernel_launch(void* const* d_in, const int* in_sizes, int n_in,
                              void* d_out, int out_size, void* d_ws, size_t ws_size,
                              hipStream_t stream) {
  (void)in_sizes; (void)n_in; (void)out_size; (void)ws_size;

  // _SPECS order: (w_input_index, co, ci)
  static const int sw[16] = {1, 3, 5, 7, 9, 11, 13, 15, 17, 19, 21, 23, 25, 27, 29, 31};
  static const int sco[16] = {32, 32, 32, 32, 32, 128, 128, 128, 128, 128,
                              128, 1024, 512, 256, 128, 13};
  static const int sci[16] = {10, 32, 6, 32, 32, 10, 128, 64, 128, 128,
                              128, 128, 1184, 512, 256, 128};

  char* ws = (char*)d_ws;
  size_t off = 0;
  auto arena = [&](size_t bytes) {
    size_t o = off;
    off = (off + bytes + 255) & ~(size_t)255;
    return o;
  };

  size_t woff[16];
  int cop[16], cip[16];
  for (int s = 0; s < 16; ++s) {
    cop[s] = (sco[s] + 15) & ~15;
    cip[s] = (sci[s] + 31) & ~31;
    woff[s] = arena((size_t)cop[s] * cip[s] * 2);
  }
  const size_t o_idx1 = arena((size_t)P1 * KNB * 4);
  const size_t o_idx2 = arena((size_t)P1 * KNB * 4);
  const size_t o_x3 = arena((size_t)P1 * LD3 * 2);
  const size_t o_bufA = arena((size_t)P2 * 128 * 2);
  const size_t o_bufB = arena((size_t)P2 * 128 * 2);
  const size_t o_bufC = arena((size_t)P2 * 128 * 2);
  const size_t o_part = arena((size_t)NBATCH * 8 * 1024 * 4);

  const float* xyz = (const float*)d_in[0];
  int* idx1 = (int*)(ws + o_idx1);
  int* idx2 = (int*)(ws + o_idx2);
  u16* X3 = (u16*)(ws + o_x3);
  u16* A = (u16*)(ws + o_bufA);
  u16* Bb = (u16*)(ws + o_bufB);
  u16* C = (u16*)(ws + o_bufC);
  float* part = (float*)(ws + o_part);
  float* outp = (float*)d_out;

  // --- 0: convert weights fp32 -> padded bf16 -------------------------------
  for (int s = 0; s < 16; ++s) {
    const int total = cop[s] * cip[s];
    cvt_w<<<(total + 255) / 256, 256, 0, stream>>>(
        (const float*)d_in[sw[s]], (u16*)(ws + woff[s]), sco[s], sci[s], cip[s], total);
  }

  auto W = [&](int s) { return (const u16*)(ws + woff[s]); };
  auto Bias = [&](int s) { return (const float*)d_in[sw[s] + 1]; };
  auto gemm = [&](int s, const u16* X, int ldx, u16* Y, int ldy, float* Yf,
                  int P, int relu) {
    dim3 g(P / 512, cop[s] / 16);
    gemm_bf16<<<g, 256, 0, stream>>>(W(s), Bias(s), X, ldx, Y, ldy, Yf,
                                     sco[s], cip[s], relu);
  };

  const dim3 knn_grid(NPT / 256, NBATCH);
  const int bld = P2 / 256;

  // --- edgeconv 1 (C=3 -> 32) ----------------------------------------------
  knn_xyz<<<knn_grid, 256, 0, stream>>>(xyz, idx1);
  build_g<<<bld, 256, 0, stream>>>(xyz, idx1, A);
  gemm(0, A, 32, Bb, 32, nullptr, P2, 1);     // wg1: A -> B
  gemm(1, Bb, 32, A, 32, nullptr, P2, 1);     // wg2: B -> A   (g final)
  build_f_xyz<<<bld, 256, 0, stream>>>(xyz, idx1, Bb);
  gemm(2, Bb, 32, C, 32, nullptr, P2, 1);     // wf1: B -> C
  gemm(3, C, 32, Bb, 32, nullptr, P2, 1);     // wf2: C -> B
  gemm(4, Bb, 32, C, 32, nullptr, P2, 1);     // wf3: B -> C   (f final)
  ec_mul_max<<<P1 * 32 / 256, 256, 0, stream>>>(A, C, 32, 32, X3, 0);

  // --- edgeconv 2 (C=32 -> 128) --------------------------------------------
  knn_fea<<<knn_grid, 256, 0, stream>>>(X3, idx2);
  build_g<<<bld, 256, 0, stream>>>(xyz, idx2, A);
  gemm(5, A, 32, Bb, 128, nullptr, P2, 1);    // wg1: A -> B
  gemm(6, Bb, 128, A, 128, nullptr, P2, 1);   // wg2: B -> A   (g final)
  build_f_fea<<<bld, 256, 0, stream>>>(X3, idx2, Bb);
  gemm(7, Bb, 64, C, 128, nullptr, P2, 1);    // wf1: B -> C
  gemm(8, C, 128, Bb, 128, nullptr, P2, 1);   // wf2: C -> B
  gemm(9, Bb, 128, C, 128, nullptr, P2, 1);   // wf3: B -> C   (f final)
  ec_mul_max<<<P1 * 128 / 256, 256, 0, stream>>>(A, C, 128, 128, X3, CH_F2);

  // --- head ----------------------------------------------------------------
  gemm(10, X3 + CH_F2, LD3, A, 128, nullptr, P1, 1);   // c1: fea2 -> A
  gemm(11, A, 128, Bb, 1024, nullptr, P1, 1);          // c2: A -> B (1024ch)
  gmax_part<<<dim3(4, 8, NBATCH), 256, 0, stream>>>(Bb, part);
  gmax_bcast<<<dim3(4, 16, NBATCH), 256, 0, stream>>>(part, X3);
  gemm(12, X3, LD3, A, 512, nullptr, P1, 1);           // c3: concat(1184) -> A
  gemm(13, A, 512, Bb, 256, nullptr, P1, 1);           // c4: A -> B
  gemm(14, Bb, 256, A, 128, nullptr, P1, 1);           // c5: B -> A
  gemm(15, A, 128, nullptr, 0, outp, P1, 0);           // c6: A -> out (B,13,N)
}
